// Block_46282567581862
// MI455X (gfx1250) — matrix operand order repaired and checked by reading, not compiled, not run on hardware
//
#include <hip/hip_runtime.h>
#include <cstdint>
#include <cmath>

// ---------------------------------------------------------------------------
// Types for CDNA5 WMMA (wave32): v16bf A/B fragments, v8f f32 accumulator
// ---------------------------------------------------------------------------
typedef __bf16 bf16x16 __attribute__((ext_vector_type(16)));
typedef float  f32x8   __attribute__((ext_vector_type(8)));
typedef unsigned int u32x4 __attribute__((ext_vector_type(4)));
typedef int  i32x4 __attribute__((ext_vector_type(4)));
typedef int  i32x8 __attribute__((ext_vector_type(8)));

#if defined(__has_builtin)
#if __has_builtin(__builtin_amdgcn_tensor_load_to_lds)
#define HAVE_TDM 1
#endif
#endif

#define DIM 768
#define SEQ 577
#define BATCH 16
#define NHEADS 12
#define HDIM 64
#define HIDDEN 3072
#define NTOK (BATCH * SEQ)   // 9232

__device__ __forceinline__ __bf16 f2bf(float f) {
    union { float f; uint32_t u; } a; a.f = f;
    uint32_t r = a.u + 0x7FFFu + ((a.u >> 16) & 1u);   // round-to-nearest-even
    union { unsigned short s; __bf16 b; } c; c.s = (unsigned short)(r >> 16);
    return c.b;
}

__device__ __forceinline__ f32x8 wmma_bf16(bf16x16 a, bf16x16 b, f32x8 c) {
    return __builtin_amdgcn_wmma_f32_16x16x32_bf16(
        /*neg_a=*/false, a, /*neg_b=*/false, b,
        /*c_mod=*/(short)0, c, /*reuse_a=*/false, /*reuse_b=*/false);
}

// CDNA5 async global->LDS copy of 16 bytes (per-lane), tracked by ASYNCcnt.
__device__ __forceinline__ void async_copy_b128(unsigned lds_dst, const void* gsrc) {
    asm volatile("global_load_async_to_lds_b128 %0, %1, off"
                 :: "v"(lds_dst), "v"(gsrc) : "memory");
}
__device__ __forceinline__ unsigned lds_addr_u32(const void* p) {
    return (unsigned)(uintptr_t)p;   // low 32 bits of flat addr == LDS offset
}

#ifdef HAVE_TDM
// TDM: DMA a rows x 32 bf16 tile (row stride K elements) into LDS with a
// 8-bf16 pad after each row (LDS row stride 40 bf16). OOB rows return zero.
// D# layout per CDNA5 ISA 8.3/8.4. Issue from ONE wave only (TDM ignores EXEC).
// This toolchain declares the 6-arg builtin:
//   (uint32x4 g0, int32x8 g1, int32x4 g2, int32x4 g3, int32x8, i32 cpol)
__device__ __forceinline__ void tdm_load_tile_a(const __bf16* gsrc, unsigned lds,
                                                unsigned rows, unsigned K) {
    unsigned long long ga = (unsigned long long)(uintptr_t)gsrc;
    u32x4 g0;
    g0[0] = 1u;                                            // count=1, user mode
    g0[1] = lds;                                           // lds_addr (bytes)
    g0[2] = (unsigned)(ga & 0xFFFFFFFFu);                  // global_addr[31:0]
    g0[3] = (unsigned)((ga >> 32) & 0x01FFFFFFu) | 0x80000000u;  // [56:32] | type=2
    i32x8 g1;
    //  data_size=1 (2B) | pad_enable | pad_interval=3 (16 DW) | pad_amount=3 (4 DW)
    g1[0] = (int)((1u << 16) | (1u << 20) | (3u << 22) | (3u << 25));
    g1[1] = (int)((K & 0xFFFFu) << 16);                    // tensor_dim0 lo
    g1[2] = (int)(((K >> 16) & 0xFFFFu) | ((rows & 0xFFFFu) << 16));   // dim0 hi | dim1 lo
    g1[3] = (int)(((rows >> 16) & 0xFFFFu) | (32u << 16)); // dim1 hi | tile_dim0=32
    g1[4] = 128;                                           // tile_dim1=128 rows
    g1[5] = (int)K;                                        // tensor_dim0_stride lo
    g1[6] = 0;
    g1[7] = 0;
    i32x4 z4 = {0, 0, 0, 0};
    i32x8 z8 = {0, 0, 0, 0, 0, 0, 0, 0};
    __builtin_amdgcn_tensor_load_to_lds(g0, g1, z4, z4, z8, 0);
}
#endif

// A fragment (16x32 bf16, 16-bit A layout): lane m = lane%16, kb = (lane/16)*8,
// elements 0..7 = K kb..kb+7, elements 8..15 = K kb+16..kb+23.
__device__ __forceinline__ bf16x16 load_a_frag(const __bf16* base, int row0, int ldm, int koff) {
    int lane = threadIdx.x & 31;
    const __bf16* p = base + (size_t)(row0 + (lane & 15)) * ldm + koff + ((lane >> 4) * 8);
    union { u32x4 q[2]; bf16x16 v; } u;
    u.q[0] = *(const u32x4*)(p);
    u.q[1] = *(const u32x4*)(p + 16);
    return u.v;
}

// B fragment (32x16 bf16) read from an N-major (transposed) tile Bt[n][k]:
// lane n = lane%16, kb = (lane/16)*8, elements 0..7 = K kb..kb+7, elements 8..15 = K kb+16..kb+23.
__device__ __forceinline__ bf16x16 load_b_frag(const __bf16* baseT, int col0, int ldn, int koff) {
    int lane = threadIdx.x & 31;
    const __bf16* p = baseT + (size_t)(col0 + (lane & 15)) * ldn + koff + ((lane >> 4) * 8);
    union { u32x4 q[2]; bf16x16 v; } u;
    u.q[0] = *(const u32x4*)(p);
    u.q[1] = *(const u32x4*)(p + 16);
    return u.v;
}

// ---------------------------------------------------------------------------
// f32 -> bf16 elementwise convert (weights)
// ---------------------------------------------------------------------------
__global__ __launch_bounds__(256) void cvt_bf16_kernel(const float* __restrict__ in,
                                                       __bf16* __restrict__ out, int n) {
    int i = blockIdx.x * 256 + threadIdx.x;
    if (i < n) out[i] = f2bf(in[i]);
}

// ---------------------------------------------------------------------------
// LayerNorm over C=768, one row per block (256 threads, 3 elems/thread), bf16 out
// ---------------------------------------------------------------------------
__global__ __launch_bounds__(256) void ln_bf16_kernel(const float* __restrict__ x,
                                                      const float* __restrict__ g,
                                                      const float* __restrict__ bias,
                                                      __bf16* __restrict__ out) {
    __shared__ float sbuf[256];
    int row = blockIdx.x;
    int tid = threadIdx.x;
    const float* xr = x + (size_t)row * DIM;
    float v[3], s = 0.f, s2 = 0.f;
#pragma unroll
    for (int i = 0; i < 3; ++i) { v[i] = xr[tid + i * 256]; s += v[i]; s2 += v[i] * v[i]; }

    sbuf[tid] = s; __syncthreads();
#pragma unroll
    for (int st = 128; st > 0; st >>= 1) { if (tid < st) sbuf[tid] += sbuf[tid + st]; __syncthreads(); }
    float tot = sbuf[0]; __syncthreads();
    sbuf[tid] = s2; __syncthreads();
#pragma unroll
    for (int st = 128; st > 0; st >>= 1) { if (tid < st) sbuf[tid] += sbuf[tid + st]; __syncthreads(); }
    float tot2 = sbuf[0];

    float mu = tot * (1.0f / DIM);
    float var = tot2 * (1.0f / DIM) - mu * mu;
    float rs = rsqrtf(var + 1e-5f);
    __bf16* orow = out + (size_t)row * DIM;
#pragma unroll
    for (int i = 0; i < 3; ++i) {
        int c = tid + i * 256;
        orow[c] = f2bf((v[i] - mu) * rs * g[c] + bias[c]);
    }
}

// ---------------------------------------------------------------------------
// Tiled bf16 GEMM: out = act(A[MxK] * W[KxN] + bias [+ res]).
//   block tile 128x128, wave tile 32x64 (2x4 wmma frags), K step 32.
//   A tiles staged via TDM tensor_load_to_lds (TENSORcnt) when available,
//   else per-lane async global->LDS (ASYNCcnt). Double buffered.
// mode: 0 = f32 out, 1 = f32 out + res, 2 = bf16 out, 3 = bf16 gelu(out)
// ---------------------------------------------------------------------------
__global__ __launch_bounds__(256) void gemm_bf16_kernel(
    const __bf16* __restrict__ A, const __bf16* __restrict__ W,
    const float* __restrict__ bias, const float* __restrict__ res,
    float* __restrict__ outf, __bf16* __restrict__ outb,
    int M, int N, int K, int mode) {
    __shared__ __bf16 As[2][128 * 40];   // A tile, row-major, padded stride 40
    __shared__ __bf16 Bs[2][128 * 40];   // W tile, n-major (transposed), stride 40

    const int m0 = blockIdx.y * 128;
    const int n0 = blockIdx.x * 128;
    const int tid = threadIdx.x;
    const int wave = tid >> 5, lane = tid & 31;
    const int wm = (wave & 3) * 32;       // wave row offset in tile
    const int wn = (wave >> 2) * 64;      // wave col offset in tile

    // stage A tile: 128 rows x 32 cols of A starting at (m0, k0)
    auto stageA = [&](int k0, int bufi) {
#ifdef HAVE_TDM
        if (wave == 0) {   // wave-uniform branch: exactly one TDM issue per stage
            tdm_load_tile_a(A + (size_t)m0 * K + k0, lds_addr_u32(&As[bufi][0]),
                            (unsigned)(M - m0), (unsigned)K);
        }
#else
#pragma unroll
        for (int i = 0; i < 2; ++i) {
            int ch = tid + i * 256;
            int r = ch >> 2;
            int cc = (ch & 3) * 8;
            int rg = m0 + r; if (rg >= M) rg = M - 1;   // clamp: OOB rows never stored
            async_copy_b128(lds_addr_u32(&As[bufi][r * 40 + cc]),
                            A + (size_t)rg * K + k0 + cc);
        }
#endif
    };
    // stage W tile transposed: 32 k-rows x 128 n-cols, 2 x 16B per thread
    auto stageB = [&](int k0, int bufi) {
#pragma unroll
        for (int i = 0; i < 2; ++i) {
            int ch = tid + i * 256;
            int k = ch >> 4;
            int nc = (ch & 15) * 8;
            union { u32x4 q; __bf16 h[8]; } val;
            val.q = *(const u32x4*)(W + (size_t)(k0 + k) * N + n0 + nc);
#pragma unroll
            for (int j = 0; j < 8; ++j) Bs[bufi][(nc + j) * 40 + k] = val.h[j];
        }
    };

    f32x8 acc[2][4];
#pragma unroll
    for (int t = 0; t < 2; ++t)
#pragma unroll
        for (int c = 0; c < 4; ++c)
#pragma unroll
            for (int r = 0; r < 8; ++r) acc[t][c][r] = 0.f;

    const int nk = K >> 5;
    stageA(0, 0);
    stageB(0, 0);
    for (int i = 0; i < nk; ++i) {
        const int cur = i & 1;
        if (i + 1 < nk) {
            stageA((i + 1) * 32, cur ^ 1);
            stageB((i + 1) * 32, cur ^ 1);
#ifdef HAVE_TDM
            if (wave == 0) __builtin_amdgcn_s_wait_tensorcnt(1);  // stage i done (in-order)
#else
            asm volatile("s_wait_asynccnt 0x2" ::: "memory");
#endif
        } else {
#ifdef HAVE_TDM
            if (wave == 0) __builtin_amdgcn_s_wait_tensorcnt(0);
#else
            asm volatile("s_wait_asynccnt 0x0" ::: "memory");
#endif
        }
        __syncthreads();

        bf16x16 aF[2], bF[4];
#pragma unroll
        for (int t = 0; t < 2; ++t) aF[t] = load_a_frag(As[cur], wm + t * 16, 40, 0);
#pragma unroll
        for (int c = 0; c < 4; ++c) bF[c] = load_b_frag(Bs[cur], wn + c * 16, 40, 0);
#pragma unroll
        for (int t = 0; t < 2; ++t)
#pragma unroll
            for (int c = 0; c < 4; ++c) acc[t][c] = wmma_bf16(aF[t], bF[c], acc[t][c]);
        __syncthreads();   // all waves done reading buf before it is restaged
    }

    // --- epilogue (C layout: lane n = lane%16, row = r + 8*(lane/16)) ---
#pragma unroll
    for (int t = 0; t < 2; ++t) {
#pragma unroll
        for (int c = 0; c < 4; ++c) {
            int col = n0 + wn + c * 16 + (lane & 15);
            float bv = bias[col];
            int mrow0 = m0 + wm + t * 16 + ((lane >> 4) * 8);
#pragma unroll
            for (int r = 0; r < 8; ++r) {
                int row = mrow0 + r;
                if (row >= M) continue;
                float v = acc[t][c][r] + bv;
                size_t idx = (size_t)row * N + col;
                if (mode == 1) {
                    outf[idx] = v + res[idx];
                } else if (mode == 2) {
                    outb[idx] = f2bf(v);
                } else if (mode == 3) {
                    outb[idx] = f2bf(0.5f * v * (1.0f + erff(v * 0.70710678118f)));
                } else {
                    outf[idx] = v;
                }
            }
        }
    }
}

// ---------------------------------------------------------------------------
// Flash attention: qkv bf16 [NTOK, 2304] (q|k|v, per-head h*64+d inside 768).
// grid (5, 12, 16): 128 queries / block, one wave owns 16 queries.
// K tile staged with async global->LDS; V transposed through VGPRs.
// ---------------------------------------------------------------------------
__global__ __launch_bounds__(256) void attn_kernel(const __bf16* __restrict__ qkv,
                                                   __bf16* __restrict__ out) {
    const int QKVLD = 3 * DIM;
    __shared__ __bf16 Kt[32 * 72];        // K tile [key][d], stride 72
    __shared__ __bf16 Vt[64 * 40];        // V tile transposed [d][key], stride 40
    __shared__ __bf16 Pa[8 * 16 * 40];    // per-wave P scratch [16][stride 40]

    const int b = blockIdx.z, h = blockIdx.y, qblk = blockIdx.x;
    const int tid = threadIdx.x, wave = tid >> 5, lane = tid & 31;
    const int qbase = qblk * 128 + wave * 16;

    const __bf16* base = qkv + (size_t)b * SEQ * QKVLD + h * HDIM;

    // --- load per-wave Q A-fragments (d 0..31 and 32..63) ---
    int mrow = qbase + (lane & 15);
    int mc = mrow < SEQ ? mrow : SEQ - 1;
    int kb = (lane >> 4) * 8;
    const __bf16* qp = base + (size_t)mc * QKVLD + kb;
    bf16x16 qA0, qA1;
    {
        union { u32x4 q[2]; bf16x16 v; } u;
        u.q[0] = *(const u32x4*)(qp);
        u.q[1] = *(const u32x4*)(qp + 16);
        qA0 = u.v;
        u.q[0] = *(const u32x4*)(qp + 32);
        u.q[1] = *(const u32x4*)(qp + 48);
        qA1 = u.v;
    }

    float m_i[8], l_i[8], corr[8];
    f32x8 O[4];
#pragma unroll
    for (int r = 0; r < 8; ++r) { m_i[r] = -1e30f; l_i[r] = 0.f; }
#pragma unroll
    for (int t = 0; t < 4; ++t)
#pragma unroll
        for (int r = 0; r < 8; ++r) O[t][r] = 0.f;

    const int NKB = (SEQ + 31) / 32;      // 19
    for (int j = 0; j < NKB; ++j) {
        const int kbj = j * 32;
        // --- stage K (async) + V (transpose), clamp OOB keys (masked below) ---
        {
            int key = tid >> 3;
            int dc = (tid & 7) * 8;
            int kg = kbj + key; if (kg >= SEQ) kg = SEQ - 1;
            async_copy_b128(lds_addr_u32(Kt + key * 72 + dc),
                            base + DIM + (size_t)kg * QKVLD + dc);
            union { u32x4 q; __bf16 h8[8]; } vv;
            vv.q = *(const u32x4*)(base + 2 * DIM + (size_t)kg * QKVLD + dc);
#pragma unroll
            for (int i = 0; i < 8; ++i) Vt[(dc + i) * 40 + key] = vv.h8[i];
        }
        asm volatile("s_wait_asynccnt 0x0" ::: "memory");
        __syncthreads();

        // --- S = Q K^T (16 queries x 32 keys), two 16-key column frags ---
        f32x8 S[2];
#pragma unroll
        for (int c = 0; c < 2; ++c) {
#pragma unroll
            for (int r = 0; r < 8; ++r) S[c][r] = 0.f;
            S[c] = wmma_bf16(qA0, load_b_frag(Kt, c * 16, 72, 0), S[c]);
            S[c] = wmma_bf16(qA1, load_b_frag(Kt, c * 16, 72, 32), S[c]);
        }

        // --- scale + mask + online softmax (rows live in own 16-lane half) ---
        float sv[2][8];
#pragma unroll
        for (int c = 0; c < 2; ++c) {
            int key = kbj + c * 16 + (lane & 15);
            bool valid = key < SEQ;
#pragma unroll
            for (int r = 0; r < 8; ++r) sv[c][r] = valid ? S[c][r] * 0.125f : -1e30f;
        }
#pragma unroll
        for (int r = 0; r < 8; ++r) {
            float mx = fmaxf(sv[0][r], sv[1][r]);
#pragma unroll
            for (int off = 1; off < 16; off <<= 1) mx = fmaxf(mx, __shfl_xor(mx, off, 32));
            float mn = fmaxf(m_i[r], mx);
            float p0 = __expf(sv[0][r] - mn);
            float p1 = __expf(sv[1][r] - mn);
            float rs = p0 + p1;
#pragma unroll
            for (int off = 1; off < 16; off <<= 1) rs += __shfl_xor(rs, off, 32);
            corr[r] = __expf(m_i[r] - mn);
            l_i[r] = l_i[r] * corr[r] + rs;
            m_i[r] = mn;
            sv[0][r] = p0; sv[1][r] = p1;
        }

        // --- P -> LDS (per-wave) -> A fragment ---
        __bf16* pw = Pa + wave * 16 * 40;
        int prow = (lane >> 4) * 8;
#pragma unroll
        for (int r = 0; r < 8; ++r) {
            pw[(prow + r) * 40 + (lane & 15)] = f2bf(sv[0][r]);
            pw[(prow + r) * 40 + 16 + (lane & 15)] = f2bf(sv[1][r]);
        }
        asm volatile("s_wait_dscnt 0x0" ::: "memory");   // in-wave LDS RAW fence
        bf16x16 pF = load_a_frag(pw, 0, 40, 0);

        // --- O = diag(corr) O + P V ---
#pragma unroll
        for (int t = 0; t < 4; ++t) {
            f32x8 o = O[t];
#pragma unroll
            for (int r = 0; r < 8; ++r) o[r] *= corr[r];
            O[t] = wmma_bf16(pF, load_b_frag(Vt, t * 16, 40, 0), o);
        }
        __syncthreads();
    }

    // --- normalize + store bf16 to [NTOK, 768] (token row, col h*64+d) ---
    int orow0 = qbase + (lane >> 4) * 8;
#pragma unroll
    for (int t = 0; t < 4; ++t) {
#pragma unroll
        for (int r = 0; r < 8; ++r) {
            int row = orow0 + r;
            if (row < SEQ) {
                out[((size_t)b * SEQ + row) * DIM + h * HDIM + t * 16 + (lane & 15)] =
                    f2bf(O[t][r] / l_i[r]);
            }
        }
    }
}

// ---------------------------------------------------------------------------
// Host launcher
// ---------------------------------------------------------------------------
extern "C" void kernel_launch(void* const* d_in, const int* in_sizes, int n_in,
                              void* d_out, int out_size, void* d_ws, size_t ws_size,
                              hipStream_t stream) {
    const float* x      = (const float*)d_in[0];
    const float* ln1_g  = (const float*)d_in[1];
    const float* ln1_b  = (const float*)d_in[2];
    const float* w_qkv  = (const float*)d_in[3];
    const float* b_qkv  = (const float*)d_in[4];
    const float* w_proj = (const float*)d_in[5];
    const float* b_proj = (const float*)d_in[6];
    const float* ln2_g  = (const float*)d_in[7];
    const float* ln2_b  = (const float*)d_in[8];
    const float* w_fc1  = (const float*)d_in[9];
    const float* b_fc1  = (const float*)d_in[10];
    const float* w_fc2  = (const float*)d_in[11];
    const float* b_fc2  = (const float*)d_in[12];
    float* outp = (float*)d_out;

    char* w = (char*)d_ws;
    size_t off = 0;
    auto take = [&](size_t bytes) -> void* {
        void* p = w + off;
        off = (off + bytes + 255) & ~(size_t)255;
        return p;
    };
    __bf16* wqkv_b  = (__bf16*)take((size_t)DIM * 3 * DIM * 2);
    __bf16* wproj_b = (__bf16*)take((size_t)DIM * DIM * 2);
    __bf16* wfc1_b  = (__bf16*)take((size_t)DIM * HIDDEN * 2);
    __bf16* wfc2_b  = (__bf16*)take((size_t)HIDDEN * DIM * 2);
    __bf16* tA      = (__bf16*)take((size_t)NTOK * DIM * 2);      // ln1 out
    __bf16* qkvb    = (__bf16*)take((size_t)NTOK * 3 * DIM * 2);  // qkv
    __bf16* attno   = (__bf16*)take((size_t)NTOK * DIM * 2);      // attention out
    float*  x1      = (float*) take((size_t)NTOK * DIM * 4);      // residual 1
    __bf16* ln2o    = (__bf16*)take((size_t)NTOK * DIM * 2);      // ln2 out
    __bf16* h1      = (__bf16*)take((size_t)NTOK * HIDDEN * 2);   // fc1 out

    auto cvt = [&](const float* src, __bf16* dst, int n) {
        cvt_bf16_kernel<<<(n + 255) / 256, 256, 0, stream>>>(src, dst, n);
    };
    cvt(w_qkv,  wqkv_b,  DIM * 3 * DIM);
    cvt(w_proj, wproj_b, DIM * DIM);
    cvt(w_fc1,  wfc1_b,  DIM * HIDDEN);
    cvt(w_fc2,  wfc2_b,  HIDDEN * DIM);

    const int M = NTOK;
    const int MT = (M + 127) / 128;   // 73

    // LN1
    ln_bf16_kernel<<<M, 256, 0, stream>>>(x, ln1_g, ln1_b, tA);
    // QKV: [M,768] x [768,2304] -> bf16
    gemm_bf16_kernel<<<dim3(3 * DIM / 128, MT), 256, 0, stream>>>(
        tA, wqkv_b, b_qkv, nullptr, nullptr, qkvb, M, 3 * DIM, DIM, 2);
    // Attention
    attn_kernel<<<dim3((SEQ + 127) / 128, NHEADS, BATCH), 256, 0, stream>>>(qkvb, attno);
    // Proj + residual -> x1 (f32)
    gemm_bf16_kernel<<<dim3(DIM / 128, MT), 256, 0, stream>>>(
        attno, wproj_b, b_proj, x, x1, nullptr, M, DIM, DIM, 1);
    // LN2
    ln_bf16_kernel<<<M, 256, 0, stream>>>(x1, ln2_g, ln2_b, ln2o);
    // FC1 + exact GELU -> bf16
    gemm_bf16_kernel<<<dim3(HIDDEN / 128, MT), 256, 0, stream>>>(
        ln2o, wfc1_b, b_fc1, nullptr, nullptr, h1, M, HIDDEN, DIM, 3);
    // FC2 + residual -> d_out (f32)
    gemm_bf16_kernel<<<dim3(DIM / 128, MT), 256, 0, stream>>>(
        h1, wfc2_b, b_fc2, x1, outp, nullptr, M, DIM, HIDDEN, 1);
}
